// OfflineSlidingWindowAttn_55147380081133
// MI455X (gfx1250) — matrix operand order repaired and checked by reading, not compiled, not run on hardware
//
#include <hip/hip_runtime.h>
#include <hip/hip_bf16.h>
#include <math.h>

#define HEAD_DIM 128
#define HQ       32
#define HKV      8
#define WINDOW   512
#define B_       2
#define S_       2048
#define WPB      8        // waves per block (attention kernel)

#define CAP      30.0f
#define EPS      1e-5f
#define SCALE_F  0.08838834764831845f   // 1/sqrt(128)
#define CLIP_A   1.06f                  // CLIP_HI - CLIP_LO
#define CLIP_B  -0.03f                  // CLIP_LO

typedef __attribute__((ext_vector_type(16))) _Float16 v16h;
typedef __attribute__((ext_vector_type(8)))  _Float16 v8h;
typedef __attribute__((ext_vector_type(4)))  _Float16 v4h;
typedef __attribute__((ext_vector_type(8)))  float    v8f;

// exp(CAP*tanh(s*SCALE/CAP) - CAP) computed branch-free via
// CAP*tanh(u) - CAP = -2*CAP / (exp(2u) + 1).
// Safe for any finite s: eu->inf => rcp->0 => result 1 (zeroed by mask later).
__device__ __forceinline__ float capped_exp(float s) {
  const float TWO_U = 2.0f * SCALE_F / CAP;   // 2u = s * TWO_U
  float eu = __expf(s * TWO_U);               // v_exp_f32 (+ mul by log2e)
  float inv = __builtin_amdgcn_rcpf(eu + 1.0f);
  return __expf(-2.0f * CAP * inv);           // exp(capped - CAP) in (0, 1]
}

// ---------------------------------------------------------------------------
// Pre-pass 1: GroupRMSNorm(q)->qn(f16), GroupRMSNorm(k)->kn(f16).
// One wave per 128-element head-row; 4 elements per lane; shuffle reduction.
// ---------------------------------------------------------------------------
__global__ __launch_bounds__(256) void norm_qk_kernel(
    const float* __restrict__ q, const float* __restrict__ k,
    const float* __restrict__ qw, const float* __restrict__ kw,
    _Float16* __restrict__ qn, _Float16* __restrict__ kn) {
  const int wv = threadIdx.x >> 5;
  const int lane = threadIdx.x & 31;
  const long gid = (long)blockIdx.x * WPB + wv;
  const long NQ = (long)B_ * S_ * HQ;
  const long NK = (long)B_ * S_ * HKV;

  const float* src;
  const float* w;
  _Float16* dst;
  if (gid < NQ) {
    src = q + gid * HEAD_DIM;
    w   = qw + (long)(gid % HQ) * HEAD_DIM;
    dst = qn + gid * HEAD_DIM;
  } else {
    long g2 = gid - NQ;
    if (g2 >= NK) return;
    src = k + g2 * HEAD_DIM;
    w   = kw + (long)(g2 % HKV) * HEAD_DIM;
    dst = kn + g2 * HEAD_DIM;
  }

  float4 x = ((const float4*)src)[lane];
  float ss = x.x * x.x + x.y * x.y + x.z * x.z + x.w * x.w;
#pragma unroll
  for (int off = 1; off < 32; off <<= 1) ss += __shfl_xor(ss, off);
  const float inv = rsqrtf(ss * (1.0f / HEAD_DIM) + EPS);

  float4 ww = ((const float4*)w)[lane];
  v4h o;
  o[0] = (_Float16)(x.x * inv * ww.x);
  o[1] = (_Float16)(x.y * inv * ww.y);
  o[2] = (_Float16)(x.z * inv * ww.z);
  o[3] = (_Float16)(x.w * inv * ww.w);
  *(v4h*)(dst + lane * 4) = o;
}

// ---------------------------------------------------------------------------
// Pre-pass 2: vT[b][kvh][d][s] = (f16) v[b][s][kvh][d]   (tiled via LDS)
// Makes PV B-fragments a single contiguous 32B load per lane.
// ---------------------------------------------------------------------------
__global__ __launch_bounds__(256) void v_transpose_kernel(
    const float* __restrict__ v, _Float16* __restrict__ vT) {
  __shared__ _Float16 tile[32][33];
  const int s0  = blockIdx.x * 32;            // S_/32 = 64 tiles
  const int kvh = blockIdx.y >> 2;            // 8 kv heads * 4 d-tiles
  const int d0  = (blockIdx.y & 3) * 32;
  const int b   = blockIdx.z;
  const int t   = threadIdx.x;

  const int di  = t & 31;
  const int si0 = t >> 5;
#pragma unroll
  for (int kk = 0; kk < 4; ++kk) {
    int si = si0 + kk * 8;
    float val = v[(((long)b * S_ + s0 + si) * HKV + kvh) * HEAD_DIM + d0 + di];
    tile[si][di] = (_Float16)val;
  }
  __syncthreads();
  const int s_i  = t & 31;
  const int d_i0 = t >> 5;
#pragma unroll
  for (int kk = 0; kk < 4; ++kk) {
    int d_i = d_i0 + kk * 8;
    vT[(((long)b * HKV + kvh) * HEAD_DIM + d0 + d_i) * S_ + s0 + s_i] = tile[s_i][d_i];
  }
}

// ---------------------------------------------------------------------------
// Main attention kernel. One wave = one 16-row q tile of one head.
// Fixed softmax max = CAP (scores are tanh-capped to <= CAP), so:
//   pass A: l[row] = sum_j exp(s-CAP)      (QK WMMA + shuffles)
//   pass B: P = clamp(1.06*exp(s-CAP)/l - 0.03, 0, 1); O += P @ V  (WMMA)
// All masking is arithmetic (cndmask), no divergent control flow in hot loop.
// ---------------------------------------------------------------------------
__global__ __launch_bounds__(256) void attn_kernel(
    const _Float16* __restrict__ qn, const _Float16* __restrict__ kn,
    const _Float16* __restrict__ vT, float* __restrict__ out) {
  const int wv   = threadIdx.x >> 5;
  const int lane = threadIdx.x & 31;
  const int n    = lane & 15;     // C-layout column / A-layout row
  const int half = lane >> 4;
  const int hk   = half * 8;

  const int q0  = (blockIdx.x * WPB + wv) * 16;
  const int h   = blockIdx.y;
  const int b   = blockIdx.z;
  const int kvh = h >> 2;         // repeat_interleave(4)

  __shared__ __attribute__((aligned(32))) _Float16 pstage[WPB][16][48];

  // ---- Q A-fragments (4 chunks of K=32 over d=128) ----
  const _Float16* qrow = qn + (((long)b * S_ + q0 + n) * HQ + h) * HEAD_DIM;
  v16h aq[4];
#pragma unroll
  for (int c = 0; c < 4; ++c) {
    aq[c] = *(const v16h*)(qrow + c * 32 + half * 16);
  }

  const _Float16* kb0 = kn + (((long)b * S_) * HKV + kvh) * HEAD_DIM;
  const long krow_stride = (long)HKV * HEAD_DIM;
  const _Float16* vtb = vT + (((long)b * HKV + kvh) * HEAD_DIM) * S_;

  int kv_lo = q0 - WINDOW;
  if (kv_lo < 0) kv_lo = 0;
  kv_lo &= ~31;
  const int nchunk = (q0 + 16 - kv_lo + 31) >> 5;

  // loop-invariant row indices for this lane's C-layout slots
  int irow[8];
#pragma unroll
  for (int r = 0; r < 8; ++r) irow[r] = q0 + r + 8 * half;

  // ---------------- pass A: row denominators ----------------
  float lsum[8];
#pragma unroll
  for (int r = 0; r < 8; ++r) lsum[r] = 0.0f;

  for (int ch = 0; ch < nchunk; ++ch) {
    const int kv0 = kv_lo + ch * 32;
#pragma unroll
    for (int t = 0; t < 2; ++t) {
      const int j = kv0 + t * 16 + n;
      const _Float16* krow = kb0 + (long)j * krow_stride;
      // load all four K fragments first, then issue WMMAs back-to-back
      v16h bk[4];
#pragma unroll
      for (int c = 0; c < 4; ++c)
        bk[c] = *(const v16h*)(krow + c * 32 + half * 16);
      v8f s = {0.f, 0.f, 0.f, 0.f, 0.f, 0.f, 0.f, 0.f};
#pragma unroll
      for (int c = 0; c < 4; ++c)
        s = __builtin_amdgcn_wmma_f32_16x16x32_f16(false, aq[c], false, bk[c],
                                                   (short)0, s, false, false);
#pragma unroll
      for (int r = 0; r < 8; ++r) {
        float e = capped_exp(s[r]);   // unconditional: no divergent branch
        const bool ok = (j <= irow[r]) && (j + WINDOW >= irow[r]);
        lsum[r] += e * (ok ? 1.0f : 0.0f);   // single v_cndmask on constant
      }
    }
  }
  float linv[8];
#pragma unroll
  for (int r = 0; r < 8; ++r) {
#pragma unroll
    for (int off = 1; off < 16; off <<= 1) lsum[r] += __shfl_xor(lsum[r], off);
    linv[r] = __builtin_amdgcn_rcpf(lsum[r]);  // diagonal always allowed -> lsum > 0
  }

  // ---------------- pass B: P = clip(softmax), O += P @ V ----------------
  v8f o[8];
#pragma unroll
  for (int dt = 0; dt < 8; ++dt) o[dt] = (v8f){0.f, 0.f, 0.f, 0.f, 0.f, 0.f, 0.f, 0.f};

  for (int ch = 0; ch < nchunk; ++ch) {
    const int kv0 = kv_lo + ch * 32;
#pragma unroll
    for (int t = 0; t < 2; ++t) {
      const int j = kv0 + t * 16 + n;
      const _Float16* krow = kb0 + (long)j * krow_stride;
      v16h bk[4];
#pragma unroll
      for (int c = 0; c < 4; ++c)
        bk[c] = *(const v16h*)(krow + c * 32 + half * 16);
      v8f s = {0.f, 0.f, 0.f, 0.f, 0.f, 0.f, 0.f, 0.f};
#pragma unroll
      for (int c = 0; c < 4; ++c)
        s = __builtin_amdgcn_wmma_f32_16x16x32_f16(false, aq[c], false, bk[c],
                                                   (short)0, s, false, false);
#pragma unroll
      for (int r = 0; r < 8; ++r) {
        float e = capped_exp(s[r]);   // unconditional
        float p = CLIP_A * e * linv[r] + CLIP_B;
        p = fminf(fmaxf(p, 0.0f), 1.0f);
        const bool ok = (j <= irow[r]) && (j + WINDOW >= irow[r]);
        p = ok ? p : 0.0f;            // cheap cndmask after clamp
        pstage[wv][r + 8 * half][t * 16 + n] = (_Float16)p;
      }
    }
    // per-wave LDS ordering: writes above -> reads below (no cross-wave dep)
    asm volatile("s_wait_dscnt 0" ::: "memory");

    v16h ap = *(const v16h*)(&pstage[wv][n][half * 16]);
    // PV: 8 d-tiles in two groups of 4 (load 4 frags, then 4 WMMAs)
#pragma unroll
    for (int g = 0; g < 2; ++g) {
      v16h bv[4];
#pragma unroll
      for (int u = 0; u < 4; ++u) {
        const int dt = g * 4 + u;
        const _Float16* vrow = vtb + (long)(dt * 16 + n) * S_;
        bv[u] = *(const v16h*)(vrow + kv0 + half * 16);
      }
#pragma unroll
      for (int u = 0; u < 4; ++u) {
        const int dt = g * 4 + u;
        o[dt] = __builtin_amdgcn_wmma_f32_16x16x32_f16(false, ap, false, bv[u],
                                                       (short)0, o[dt], false, false);
      }
    }
    asm volatile("s_wait_dscnt 0" ::: "memory");  // ap consumed before next chunk's stores
  }

  // ---------------- epilogue: store O (f32) ----------------
#pragma unroll
  for (int dt = 0; dt < 8; ++dt) {
#pragma unroll
    for (int r = 0; r < 8; ++r) {
      out[(((long)b * S_ + q0 + r + 8 * half) * HQ + h) * HEAD_DIM + dt * 16 + n] =
          o[dt][r];
    }
  }
}

// ---------------------------------------------------------------------------
extern "C" void kernel_launch(void* const* d_in, const int* in_sizes, int n_in,
                              void* d_out, int out_size, void* d_ws, size_t ws_size,
                              hipStream_t stream) {
  const float* q  = (const float*)d_in[0];
  const float* k  = (const float*)d_in[1];
  const float* v  = (const float*)d_in[2];
  const float* qw = (const float*)d_in[3];
  const float* kw = (const float*)d_in[4];
  float* out = (float*)d_out;

  _Float16* qn = (_Float16*)d_ws;                       // B*S*HQ*D  halves
  _Float16* kn = qn + (long)B_ * S_ * HQ * HEAD_DIM;    // B*S*HKV*D halves
  _Float16* vT = kn + (long)B_ * S_ * HKV * HEAD_DIM;   // B*HKV*D*S halves

  // Pre-pass 1: fused GroupRMSNorm + f16 convert for Q and K
  {
    long rows = (long)B_ * S_ * (HQ + HKV);
    dim3 grid((unsigned)((rows + WPB - 1) / WPB));
    norm_qk_kernel<<<grid, 256, 0, stream>>>(q, k, qw, kw, qn, kn);
  }
  // Pre-pass 2: V -> vT (f16, d-major)
  {
    dim3 grid(S_ / 32, HKV * 4, B_);
    v_transpose_kernel<<<grid, 256, 0, stream>>>(v, vT);
  }
  // Main attention
  {
    dim3 grid(S_ / (16 * WPB), HQ, B_);
    attn_kernel<<<grid, WPB * 32, 0, stream>>>(qn, kn, vT, out);
  }
}
